// HyperSphere_56453050139330
// MI455X (gfx1250) — matrix operand order repaired and checked by reading, not compiled, not run on hardware
//
#include <hip/hip_runtime.h>
#include <hip/hip_bf16.h>
#include <math.h>

typedef _Float16 v8h  __attribute__((ext_vector_type(8)));
typedef _Float16 v16h __attribute__((ext_vector_type(16)));
typedef float    v8f  __attribute__((ext_vector_type(8)));

#define NPTS   32768
#define KCODE  4096
#define DIM    18
#define DPAD   32
#define HW     4096          // 64*64
#define NBLK   128
#define PTS_PER_BLOCK (NPTS / NBLK)          // 256
#define TILES_PER_BLOCK (PTS_PER_BLOCK / 16) // 16
#define NWAVE  8
#define COLS_PER_WAVE (KCODE / NWAVE)        // 512
#define CB_PER_WAVE (COLS_PER_WAVE / 16)     // 32
#define LREF   100.0f                         // logits = 100*sim, max possible = 100
#define K2     144.2695040888963f             // 100 * log2(e): exp(100(s-1)) = 2^(K2*s - K2)

// raw v_exp_f32 (2^x), no subnormal guard: args < -126 flush to 0,
// i.e. softmax terms more than ~e^-87 below the max vanish -- same as
// what a max-subtracted f32 softmax produces.
#define EXP2_RAW(x) __builtin_amdgcn_exp2f(x)

// ---------------- prep kernels ----------------

__global__ void vq_init(float* acc, int n) {
    int i = blockIdx.x * 256 + threadIdx.x;
    if (i < n) acc[i] = 0.f;
}

__global__ void vq_prep_x(const float* __restrict__ x, _Float16* __restrict__ xnh) {
    int p = blockIdx.x * 256 + threadIdx.x;
    if (p >= NPTS) return;
    int bb = p >> 12, hw = p & (HW - 1);
    const float* xb = x + (size_t)bb * DIM * HW + hw;
    float v[DIM]; float ss = 0.f;
    #pragma unroll
    for (int dd = 0; dd < DIM; ++dd) { v[dd] = xb[(size_t)dd * HW]; ss += v[dd] * v[dd]; }
    float inv = rsqrtf(ss);
    _Float16* o = xnh + (size_t)p * DPAD;
    #pragma unroll
    for (int dd = 0; dd < DIM; ++dd) o[dd] = (_Float16)(v[dd] * inv);
    #pragma unroll
    for (int dd = DIM; dd < DPAD; ++dd) o[dd] = (_Float16)0.f;
}

__global__ void vq_prep_sp(const float* __restrict__ sp, _Float16* __restrict__ sph) {
    int k = blockIdx.x * 256 + threadIdx.x;
    if (k >= KCODE) return;
    const float* s = sp + (size_t)k * DIM;
    _Float16* o = sph + (size_t)k * DPAD;
    #pragma unroll
    for (int dd = 0; dd < DIM; ++dd) o[dd] = (_Float16)s[dd];
    #pragma unroll
    for (int dd = DIM; dd < DPAD; ++dd) o[dd] = (_Float16)0.f;
}

// ---------------- main kernel ----------------

__global__ __launch_bounds__(256) void vq_main(
    const _Float16* __restrict__ xnh,
    const _Float16* __restrict__ sph,
    const float*    __restrict__ sp,      // f32 codebook for the gather
    float* __restrict__ q_out,            // [8*18*4096]
    float* __restrict__ idx_out,          // [32768] index stored as float
    float* __restrict__ avg_acc,          // [4096] global prob accumulator
    float* __restrict__ scal_acc)         // [0]=entropy sum, [1]=commit sum
{
    __shared__ float s_avg[KCODE];
    __shared__ float s_z[NWAVE][16];
    __shared__ float s_s[NWAVE][16];
    __shared__ float s_mx[NWAVE][16];
    __shared__ int   s_ai[NWAVE][16];
    __shared__ float s_rz[16];            // 1/Z per row, for pass 2
    __shared__ float s_ent;
    __shared__ float s_com;

    const int tid  = threadIdx.x;
    const int lane = tid & 31;
    const int wave = tid >> 5;
    const int l16  = lane & 15;
    const int hh   = lane >> 4;     // which half of the wave

    for (int i = tid; i < KCODE; i += 256) s_avg[i] = 0.f;
    if (tid == 0) { s_ent = 0.f; s_com = 0.f; }
    __syncthreads();

    const int base_pt = blockIdx.x * PTS_PER_BLOCK;
    const int col0 = wave * COLS_PER_WAVE;

    for (int t = 0; t < TILES_PER_BLOCK; ++t) {
        const int tile0 = base_pt + t * 16;

        // ---- A fragment: 16x32 f16, row = tile0 + l16 (contiguous k = 16*hh + s, matches B) ----
        v16h A;
        {
            const _Float16* arow = xnh + (size_t)(tile0 + l16) * DPAD;
            v8h lo = *(const v8h*)(arow + hh * 16);
            v8h hi = *(const v8h*)(arow + hh * 16 + 8);
            #pragma unroll
            for (int i = 0; i < 8; ++i) { A[i] = lo[i]; A[i + 8] = hi[i]; }
        }
        // prefetch next tile's A row
        if (t + 1 < TILES_PER_BLOCK)
            __builtin_prefetch(xnh + (size_t)(tile0 + 16 + l16) * DPAD, 0, 3);

        // stats on the raw sim scale: Z = sum 2^(K2*s - K2), S = sum s * e
        float Z[8], S[8], mx[8]; int ai[8];
        #pragma unroll
        for (int j = 0; j < 8; ++j) { Z[j] = 0.f; S[j] = 0.f; mx[j] = -1e30f; ai[j] = 0; }

        // ---- pass 1: branchless stats sweep over this wave's 512 codes ----
        #pragma unroll 2
        for (int cb = 0; cb < CB_PER_WAVE; ++cb) {
            const int code = col0 + cb * 16 + l16;       // B column this lane holds
            v16h B;
            {
                const _Float16* brow = sph + (size_t)code * DPAD + hh * 16;
                v8h lo = *(const v8h*)(brow);
                v8h hi = *(const v8h*)(brow + 8);
                #pragma unroll
                for (int i = 0; i < 8; ++i) { B[i] = lo[i]; B[i + 8] = hi[i]; }
            }
            v8f C = {};
            C = __builtin_amdgcn_wmma_f32_16x16x32_f16(false, A, false, B,
                                                       (short)0, C, false, false);
            #pragma unroll
            for (int j = 0; j < 8; ++j) {
                float s = C[j];
                float e = EXP2_RAW(fmaf(s, K2, -K2));    // exp(100*(s-1)): 1 fma + v_exp
                Z[j] += e;
                S[j] = fmaf(s, e, S[j]);
                bool gt = s > mx[j];
                mx[j] = gt ? s : mx[j];
                ai[j] = gt ? code : ai[j];
            }
        }

        // ---- cross-lane reduce within each 16-lane half (plain sums) ----
        #pragma unroll
        for (int mask = 1; mask < 16; mask <<= 1) {
            #pragma unroll
            for (int j = 0; j < 8; ++j) {
                Z[j] += __shfl_xor(Z[j], mask);
                S[j] += __shfl_xor(S[j], mask);
                float x2 = __shfl_xor(mx[j], mask);
                int   a2 = __shfl_xor(ai[j], mask);
                bool take = (x2 > mx[j]) || (x2 == mx[j] && a2 < ai[j]);
                mx[j] = take ? x2 : mx[j];
                ai[j] = take ? a2 : ai[j];
            }
        }
        if (l16 == 0) {
            #pragma unroll
            for (int j = 0; j < 8; ++j) {
                int r = j + hh * 8;
                s_z[wave][r] = Z[j]; s_s[wave][r] = S[j];
                s_mx[wave][r] = mx[j]; s_ai[wave][r] = ai[j];
            }
        }
        __syncthreads();

        // ---- cross-wave merge, finalize per-point outputs (16 threads) ----
        if (tid < 16) {
            int r = tid;
            float Zt = s_z[0][r], St = s_s[0][r], X = s_mx[0][r];
            int A2 = s_ai[0][r];
            #pragma unroll
            for (int w = 1; w < NWAVE; ++w) {
                Zt += s_z[w][r]; St += s_s[w][r];
                float x2 = s_mx[w][r]; int a2 = s_ai[w][r];
                bool take = (x2 > X) || (x2 == X && a2 < A2);
                X  = take ? x2 : X;
                A2 = take ? a2 : A2;
            }
            float rz  = 1.0f / Zt;
            s_rz[r] = rz;
            float lse = LREF + __logf(Zt);
            float ent = lse - LREF * St * rz;          // sample entropy (logits = 100*sim)
            float commit = 2.f - 2.f * X;              // ||xn - q||^2 on unit sphere
            atomicAdd(&s_ent, ent);
            atomicAdd(&s_com, commit);

            int pt = tile0 + r;
            int bb = pt >> 12, hw = pt & (HW - 1);
            const float* srow = sp + (size_t)A2 * DIM;
            float* qb = q_out + (size_t)bb * DIM * HW + hw;
            #pragma unroll
            for (int dd = 0; dd < DIM; ++dd) qb[(size_t)dd * HW] = srow[dd];
            idx_out[pt] = (float)A2;
        }
        __syncthreads();

        // hoist 1/Z for this lane's 8 rows
        float rz[8];
        #pragma unroll
        for (int j = 0; j < 8; ++j) rz[j] = s_rz[j + hh * 8];

        // ---- pass 2: recompute sims, accumulate probs into LDS avg ----
        #pragma unroll 2
        for (int cb = 0; cb < CB_PER_WAVE; ++cb) {
            const int code = col0 + cb * 16 + l16;
            v16h B;
            {
                const _Float16* brow = sph + (size_t)code * DPAD + hh * 16;
                v8h lo = *(const v8h*)(brow);
                v8h hi = *(const v8h*)(brow + 8);
                #pragma unroll
                for (int i = 0; i < 8; ++i) { B[i] = lo[i]; B[i + 8] = hi[i]; }
            }
            v8f C = {};
            C = __builtin_amdgcn_wmma_f32_16x16x32_f16(false, A, false, B,
                                                       (short)0, C, false, false);
            float colsum = 0.f;
            #pragma unroll
            for (int j = 0; j < 8; ++j) {
                float e = EXP2_RAW(fmaf(C[j], K2, -K2)); // exp(l - 100)
                colsum = fmaf(e, rz[j], colsum);         // p = e / Z_row
            }
            colsum += __shfl_xor(colsum, 16);            // add other half's 8 rows
            if (hh == 0) atomicAdd(&s_avg[code], colsum);
        }
        // next tile's stats-write barrier orders s_rz reuse
    }
    __syncthreads();

    // ---- flush block-local accumulators ----
    for (int i = tid; i < KCODE; i += 256) atomicAdd(&avg_acc[i], s_avg[i]);
    if (tid == 0) {
        atomicAdd(&scal_acc[0], s_ent);
        atomicAdd(&scal_acc[1], s_com);
    }
}

// ---------------- final reduction ----------------

__global__ __launch_bounds__(256) void vq_final(const float* __restrict__ avg_acc,
                                                const float* __restrict__ scal_acc,
                                                float* __restrict__ out_scal) {
    __shared__ float red[256];
    float s = 0.f;
    for (int i = threadIdx.x; i < KCODE; i += 256) {
        float a = avg_acc[i] * (1.0f / (float)NPTS);
        s += a * __logf(a + 1e-15f);
    }
    red[threadIdx.x] = s;
    __syncthreads();
    for (int off = 128; off > 0; off >>= 1) {
        if (threadIdx.x < off) red[threadIdx.x] += red[threadIdx.x + off];
        __syncthreads();
    }
    if (threadIdx.x == 0) {
        float mean_entro = -red[0];
        float entro_mean = scal_acc[0] * (1.0f / (float)NPTS);
        float commit     = scal_acc[1] * (1.0f / (float)(NPTS * DIM));
        out_scal[0] = entro_mean;
        out_scal[1] = mean_entro;
        out_scal[2] = entro_mean - mean_entro;
        out_scal[3] = commit;
    }
}

// ---------------- launcher ----------------

extern "C" void kernel_launch(void* const* d_in, const int* in_sizes, int n_in,
                              void* d_out, int out_size, void* d_ws, size_t ws_size,
                              hipStream_t stream) {
    (void)in_sizes; (void)n_in; (void)out_size; (void)ws_size;
    const float* x  = (const float*)d_in[0];   // [8,18,64,64]
    const float* sp = (const float*)d_in[1];   // [4096,18]

    char* ws = (char*)d_ws;
    _Float16* xnh = (_Float16*)ws;                                   // 32768*32*2 = 2 MiB
    _Float16* sph = (_Float16*)(ws + (size_t)2 * 1024 * 1024);       // 4096*32*2  = 256 KiB
    float* avg_acc = (float*)(ws + (size_t)2 * 1024 * 1024 + 256 * 1024);
    float* scal_acc = avg_acc + KCODE;

    float* q_out    = (float*)d_out;                 // 589824
    float* idx_out  = q_out + (size_t)8 * DIM * HW;  // 32768
    float* out_scal = idx_out + NPTS;                // 4 scalars

    vq_init   <<<(KCODE + 2 + 255) / 256, 256, 0, stream>>>(avg_acc, KCODE + 2);
    vq_prep_x <<<NPTS / 256, 256, 0, stream>>>(x, xnh);
    vq_prep_sp<<<KCODE / 256, 256, 0, stream>>>(sp, sph);
    vq_main   <<<NBLK, 256, 0, stream>>>(xnh, sph, sp, q_out, idx_out, avg_acc, scal_acc);
    vq_final  <<<1, 256, 0, stream>>>(avg_acc, scal_acc, out_scal);
}
